// PARAModel_34359738521
// MI455X (gfx1250) — compile-verified, matrix operand order checked by reading, not run on hardware
//
#include <hip/hip_runtime.h>

// ---------------------------------------------------------------------------
// DeBERTa-v2 forward for gfx1250 (MI455X).
// bf16 WMMA GEMMs with double-buffered CDNA5 async global->LDS staging
// (ASYNCcnt software pipeline), fragment-ready LDS layout (b128 ds traffic),
// + fused row kernels for LN / disentangled softmax / GELU.
// ---------------------------------------------------------------------------

typedef __bf16 v16bf __attribute__((ext_vector_type(16)));
typedef __bf16 v8bf  __attribute__((ext_vector_type(8)));
typedef float  v8f   __attribute__((ext_vector_type(8)));

#define Bn    4
#define SEQ   512
#define VOCAB 25
#define HID   512
#define NHD   8
#define NL    12
#define SPAN  512
#define HEAD  64
#define FF    2048
#define TOK   (Bn * SEQ)   // 2048 rows

// ---------------------------- helpers --------------------------------------

__device__ __forceinline__ __bf16 f2bf(float f) {
  unsigned u = __builtin_bit_cast(unsigned, f);
  unsigned r = (u + 0x7fffu + ((u >> 16) & 1u)) >> 16;
  unsigned short s = (unsigned short)r;
  return __builtin_bit_cast(__bf16, s);
}

// Low 32 bits of a generic pointer into the LDS aperture are the LDS byte
// address (CDNA5 ISA 10.2).
__device__ __forceinline__ unsigned lds_addr32(const void* p) {
  return (unsigned)(unsigned long long)p;
}

// GLOBAL_LOAD_ASYNC_TO_LDS_*: copy memory -> LDS directly, ASYNCcnt path.
__device__ __forceinline__ void async_copy_b128(unsigned lds_addr,
                                                const void* gaddr) {
  asm volatile("global_load_async_to_lds_b128 %0, %1, off"
               :
               : "v"(lds_addr), "v"((unsigned long long)gaddr)
               : "memory");
}

__device__ __forceinline__ void async_copy_b64(unsigned lds_addr,
                                               const void* gaddr) {
  asm volatile("global_load_async_to_lds_b64 %0, %1, off"
               :
               : "v"(lds_addr), "v"((unsigned long long)gaddr)
               : "memory");
}

__device__ __forceinline__ void wait_async_le0() {
#if __has_builtin(__builtin_amdgcn_s_wait_asynccnt)
  __builtin_amdgcn_s_wait_asynccnt(0);
#else
  asm volatile("s_wait_asynccnt 0x0" ::: "memory");
#endif
}

// In-order ASYNCcnt completion: <=3 drains everything older than the 3 ops
// just issued for the next tile.
__device__ __forceinline__ void wait_async_le3() {
#if __has_builtin(__builtin_amdgcn_s_wait_asynccnt)
  __builtin_amdgcn_s_wait_asynccnt(3);
#else
  asm volatile("s_wait_asynccnt 0x3" ::: "memory");
#endif
}

// ------------------------- fp32 -> bf16 convert ----------------------------

__global__ void cvt_bf16_kernel(const float* __restrict__ in,
                                __bf16* __restrict__ out, long long n) {
  long long i = (long long)blockIdx.x * blockDim.x + threadIdx.x;
  long long stride = (long long)gridDim.x * blockDim.x;
  for (; i < n; i += stride) out[i] = f2bf(in[i]);
}

// ------------- fp32 [z][R][C] -> bf16 [z][C][R] tiled transpose ------------

__global__ __launch_bounds__(256) void transpose_f32_to_bf16(
    const float* __restrict__ in, __bf16* __restrict__ out, int R, int C) {
  const int z = blockIdx.z;
  in  += (long long)z * R * C;
  out += (long long)z * R * C;
  __shared__ float t[32][33];
  const int r0 = blockIdx.y * 32, c0 = blockIdx.x * 32;
  const int tx = threadIdx.x & 31, ty = threadIdx.x >> 5;  // ty 0..7
#pragma unroll
  for (int i = ty; i < 32; i += 8)
    t[i][tx] = in[(long long)(r0 + i) * C + (c0 + tx)];
  __syncthreads();
#pragma unroll
  for (int i = ty; i < 32; i += 8)
    out[(long long)(c0 + i) * R + (r0 + tx)] = f2bf(t[tx][i]);
}

// ------------------ bf16 [z][R][C] -> bf16 [z][C][R] transpose -------------

__global__ __launch_bounds__(256) void transpose_bf16(
    const __bf16* __restrict__ in, __bf16* __restrict__ out, int R, int C) {
  const int z = blockIdx.z;
  in  += (long long)z * R * C;
  out += (long long)z * R * C;
  __shared__ __bf16 t[32][33];
  const int r0 = blockIdx.y * 32, c0 = blockIdx.x * 32;
  const int tx = threadIdx.x & 31, ty = threadIdx.x >> 5;
#pragma unroll
  for (int i = ty; i < 32; i += 8)
    t[i][tx] = in[(long long)(r0 + i) * C + (c0 + tx)];
  __syncthreads();
#pragma unroll
  for (int i = ty; i < 32; i += 8)
    out[(long long)(c0 + i) * R + (r0 + tx)] = t[tx][i];
}

// ------------------------------ WMMA GEMM ----------------------------------
// C[M,N] = A[M,K] (bf16, K-contiguous, leading dim lda)
//        x B[N,K] (bf16, K-contiguous, leading dim ldb)  + bias[N]
// 128 threads = 4 waves; each wave owns one 16x16 tile; block covers 16x64.
// LDS tiles are fragment-ready (lane's 16 WMMA operands contiguous):
//   8-elem K-chunk c of row r -> lane r + 16*(c&1), elems 8*(c>>1)
//   4-elem K-group q of row r -> lane r + 16*((q>>1)&1), elems (q&1)*4+(q>>2)*8
// EDGE=false (N % 64 == 0): double-buffered async staging; every thread
//   issues exactly 3 async ops/tile (1x b64 A + 2x b128 B), so waves have a
//   uniform in-flight count and the pipeline waits asynccnt<=3.
// EDGE=true: register path with N-edge zero-fill (logits only).

template <bool EDGE>
__global__ __launch_bounds__(128) void gemm_bf16_wmma(
    const __bf16* __restrict__ A, const __bf16* __restrict__ Bm,
    const float* __restrict__ bias,
    float* __restrict__ Cf, __bf16* __restrict__ Cbf,
    int lda, int ldb, int ldc,
    int M, int N, int K, int zInner,
    long long sAo, long long sAi, long long sBo, long long sBi,
    long long sCo, long long sCi) {
  const int z = blockIdx.z;
  const long long offA = (long long)(z / zInner) * sAo + (long long)(z % zInner) * sAi;
  const long long offB = (long long)(z / zInner) * sBo + (long long)(z % zInner) * sBi;
  const long long offC = (long long)(z / zInner) * sCo + (long long)(z % zInner) * sCi;
  A  += offA;
  Bm += offB;

  const int tid  = threadIdx.x;
  const int lane = tid & 31;
  const int wv   = tid >> 5;          // wave 0..3
  const int ln   = lane & 15;
  const int g    = lane >> 4;
  const int row0   = blockIdx.y * 16;
  const int colBlk = blockIdx.x * 64;
  const int colW   = colBlk + wv * 16;

  __shared__ alignas(32) __bf16 Afrag[2][32][16];     // 2x 16x32 A tile
  __shared__ alignas(32) __bf16 Bfrag[2][4][32][16];  // 2x 4x (32x16 Bt tile)

  v8f acc;
  {
    const int n = colW + ln;
    const float bv = (bias != nullptr && n < N) ? bias[n] : 0.0f;
#pragma unroll
    for (int v = 0; v < 8; ++v) acc[v] = bv;
  }

  // Fixed per-thread staging coordinates.
  const int ar = tid >> 3, aq = tid & 7;                 // A b64 chunk
  const int aLane = ar + 16 * ((aq >> 1) & 1);
  const int aE = (aq & 1) * 4 + (aq >> 2) * 8;

  if (!EDGE) {
    // Issue 3 async ops (uniform per wave) staging tile at k-offset kk.
    auto stage = [&](int buf, int kk) {
      async_copy_b64(lds_addr32(&Afrag[buf][aLane][aE]),
                     &A[(long long)(row0 + ar) * lda + kk + aq * 4]);
#pragma unroll
      for (int t = 0; t < 2; ++t) {
        const int c = tid * 2 + t;
        const int nn = c >> 2, cb = c & 3;
        async_copy_b128(
            lds_addr32(&Bfrag[buf][nn >> 4][(nn & 15) + 16 * (cb & 1)]
                                           [8 * (cb >> 1)]),
            &Bm[(long long)(colBlk + nn) * ldb + kk + cb * 8]);
      }
    };

    stage(0, 0);
    int cur = 0;
    for (int k0 = 0; k0 < K; k0 += 32) {
      const int nxt = cur ^ 1;
      if (k0 + 32 < K) {
        stage(nxt, k0 + 32);  // overlap next tile's DMA with this tile's math
        if (k0 + 64 < K)
          __builtin_prefetch(&A[(long long)(row0 + ar) * lda + k0 + 64], 0, 1);
        wait_async_le3();
      } else {
        wait_async_le0();
      }
      __syncthreads();
      const v16bf af  = *reinterpret_cast<const v16bf*>(Afrag[cur][lane]);
      const v16bf bfr = *reinterpret_cast<const v16bf*>(Bfrag[cur][wv][lane]);
      acc = __builtin_amdgcn_wmma_f32_16x16x32_bf16(
          false, af, false, bfr, (short)0, acc, false, false);
      __syncthreads();  // all waves done reading `cur` before it is rewritten
      cur = nxt;
    }
  } else {
    // Register path with N-edge zero fill (single buffer).
    const int ar2 = tid >> 2, acb = tid & 3;
    for (int k0 = 0; k0 < K; k0 += 32) {
      if (tid < 64) {
        const v8bf ch = *reinterpret_cast<const v8bf*>(
            &A[(long long)(row0 + ar2) * lda + k0 + acb * 8]);
        *reinterpret_cast<v8bf*>(
            &Afrag[0][ar2 + 16 * (acb & 1)][8 * (acb >> 1)]) = ch;
      }
#pragma unroll
      for (int t = 0; t < 2; ++t) {
        const int c = tid * 2 + t;
        const int nn = c >> 2, cb = c & 3;
        const int gn = colBlk + nn;
        v8bf ch;
#pragma unroll
        for (int e = 0; e < 8; ++e) ch[e] = (__bf16)0.0f;
        if (gn < N)
          ch = *reinterpret_cast<const v8bf*>(
              &Bm[(long long)gn * ldb + k0 + cb * 8]);
        *reinterpret_cast<v8bf*>(
            &Bfrag[0][nn >> 4][(nn & 15) + 16 * (cb & 1)][8 * (cb >> 1)]) = ch;
      }
      __syncthreads();
      const v16bf af  = *reinterpret_cast<const v16bf*>(Afrag[0][lane]);
      const v16bf bfr = *reinterpret_cast<const v16bf*>(Bfrag[0][wv][lane]);
      acc = __builtin_amdgcn_wmma_f32_16x16x32_bf16(
          false, af, false, bfr, (short)0, acc, false, false);
      __syncthreads();
    }
  }

  const int n = colW + ln;
  if (n < N) {
#pragma unroll
    for (int v = 0; v < 8; ++v) {
      const long long m = row0 + v + 8 * g;
      const float val = acc[v];
      if (Cf)  Cf[offC + m * ldc + n]  = val;
      if (Cbf) Cbf[offC + m * ldc + n] = f2bf(val);
    }
  }
}

// ----------------------- embedding gather + LayerNorm ----------------------

__global__ __launch_bounds__(256) void embed_ln_kernel(
    const int* __restrict__ x, const int* __restrict__ amask,
    const float* __restrict__ wemb, const float* __restrict__ g,
    const float* __restrict__ b, float* __restrict__ h,
    __bf16* __restrict__ hbf) {
  const int row = blockIdx.x;
  const int tid = threadIdx.x;
  const long long base = (long long)row * HID;
  const int tok = x[row];
  const float v0 = wemb[(long long)tok * HID + tid];
  const float v1 = wemb[(long long)tok * HID + 256 + tid];

  __shared__ float red[256];
  red[tid] = v0 + v1;
  __syncthreads();
  for (int s = 128; s > 0; s >>= 1) {
    if (tid < s) red[tid] += red[tid + s];
    __syncthreads();
  }
  const float mean = red[0] * (1.0f / HID);
  __syncthreads();
  const float d0 = v0 - mean, d1 = v1 - mean;
  red[tid] = d0 * d0 + d1 * d1;
  __syncthreads();
  for (int s = 128; s > 0; s >>= 1) {
    if (tid < s) red[tid] += red[tid + s];
    __syncthreads();
  }
  const float inv = rsqrtf(red[0] * (1.0f / HID) + 1e-7f);
  const float mk = (float)amask[row];
  const float o0 = (d0 * inv * g[tid] + b[tid]) * mk;
  const float o1 = (d1 * inv * g[256 + tid] + b[256 + tid]) * mk;
  h[base + tid] = o0;
  h[base + 256 + tid] = o1;
  hbf[base + tid] = f2bf(o0);
  hbf[base + 256 + tid] = f2bf(o1);
}

// -------------------------- residual + LayerNorm ---------------------------

__global__ __launch_bounds__(256) void residual_ln_kernel(
    const float* __restrict__ a, const float* __restrict__ r,
    const float* __restrict__ g, const float* __restrict__ bta,
    float* __restrict__ out, __bf16* __restrict__ outbf) {
  const int row = blockIdx.x;
  const int tid = threadIdx.x;
  const long long base = (long long)row * HID;
  const float v0 = a[base + tid] + r[base + tid];
  const float v1 = a[base + 256 + tid] + r[base + 256 + tid];

  __shared__ float red[256];
  red[tid] = v0 + v1;
  __syncthreads();
  for (int s = 128; s > 0; s >>= 1) {
    if (tid < s) red[tid] += red[tid + s];
    __syncthreads();
  }
  const float mean = red[0] * (1.0f / HID);
  __syncthreads();
  const float d0 = v0 - mean, d1 = v1 - mean;
  red[tid] = d0 * d0 + d1 * d1;
  __syncthreads();
  for (int s = 128; s > 0; s >>= 1) {
    if (tid < s) red[tid] += red[tid + s];
    __syncthreads();
  }
  const float inv = rsqrtf(red[0] * (1.0f / HID) + 1e-7f);
  const float o0 = d0 * inv * g[tid] + bta[tid];
  const float o1 = d1 * inv * g[256 + tid] + bta[256 + tid];
  out[base + tid] = o0;
  out[base + 256 + tid] = o1;
  outbf[base + tid] = f2bf(o0);
  outbf[base + 256 + tid] = f2bf(o1);
}

// ------------- disentangled-attention gather + masked softmax --------------
// scores[b,h,i,j] = (qk[i,j] + qpk[i, clip(i-j+SPAN)] + kpq[j, clip(i-j+SPAN)])
//                   / sqrt(3*HEAD); XSoftmax with attention mask; probs bf16.

__global__ __launch_bounds__(256) void attn_softmax_kernel(
    const float* __restrict__ qk, const float* __restrict__ qpk,
    const float* __restrict__ kpq, const int* __restrict__ amask,
    __bf16* __restrict__ probs) {
  const int i = blockIdx.x, hd = blockIdx.y, b = blockIdx.z;
  const int tid = threadIdx.x;
  const long long bh = (long long)b * NHD + hd;
  const long long rqk = (bh * SEQ + i) * SEQ;
  const long long rqpk = (bh * SEQ + i) * (2 * SPAN);
  const long long rkpq = bh * SEQ * (2 * SPAN);
  const float NEG = -3.4028235e38f;
  const float inv_scale = 0.07216878364870323f;  // 1/sqrt(3*HEAD)
  const int mi = amask[b * SEQ + i];

  float s[2];
  int valid[2];
#pragma unroll
  for (int t = 0; t < 2; ++t) {
    const int j = tid + t * 256;
    int idx = i - j + SPAN;
    idx = idx < 0 ? 0 : (idx > 2 * SPAN - 1 ? 2 * SPAN - 1 : idx);
    const float sc =
        (qk[rqk + j] + qpk[rqpk + idx] +
         kpq[rkpq + (long long)j * (2 * SPAN) + idx]) * inv_scale;
    const int mj = amask[b * SEQ + j];
    valid[t] = (mi != 0) && (mj != 0);
    s[t] = valid[t] ? sc : NEG;
  }

  __shared__ float red[256];
  red[tid] = fmaxf(s[0], s[1]);
  __syncthreads();
  for (int st = 128; st > 0; st >>= 1) {
    if (tid < st) red[tid] = fmaxf(red[tid], red[tid + st]);
    __syncthreads();
  }
  const float rowmax = red[0];
  __syncthreads();
  const float e0 = __expf(s[0] - rowmax);
  const float e1 = __expf(s[1] - rowmax);
  red[tid] = e0 + e1;
  __syncthreads();
  for (int st = 128; st > 0; st >>= 1) {
    if (tid < st) red[tid] += red[tid + st];
    __syncthreads();
  }
  const float denom = red[0];
  const float rinv = denom > 0.0f ? 1.0f / denom : 0.0f;
  probs[rqk + tid]       = f2bf(valid[0] ? e0 * rinv : 0.0f);
  probs[rqk + tid + 256] = f2bf(valid[1] ? e1 * rinv : 0.0f);
}

// --------------------------------- GELU ------------------------------------

__global__ void gelu_bf16_kernel(const float* __restrict__ in,
                                 __bf16* __restrict__ out, long long n) {
  long long i = (long long)blockIdx.x * blockDim.x + threadIdx.x;
  const long long stride = (long long)gridDim.x * blockDim.x;
  for (; i < n; i += stride) {
    const float v = in[i];
    out[i] = f2bf(0.5f * v * (1.0f + erff(v * 0.70710678118654752f)));
  }
}

// ------------------------------ host driver --------------------------------

static void launch_gemm(hipStream_t stream, const __bf16* A, const __bf16* Bm,
                        const float* bias, float* Cf, __bf16* Cbf, int lda,
                        int ldb, int ldc, int M, int N, int K,
                        int batch, int zInner, long long sAo, long long sAi,
                        long long sBo, long long sBi, long long sCo,
                        long long sCi) {
  dim3 gr((N + 63) / 64, M / 16, batch);
  if ((N & 63) == 0) {
    gemm_bf16_wmma<false><<<gr, 128, 0, stream>>>(
        A, Bm, bias, Cf, Cbf, lda, ldb, ldc, M, N, K, zInner, sAo, sAi, sBo,
        sBi, sCo, sCi);
  } else {
    gemm_bf16_wmma<true><<<gr, 128, 0, stream>>>(
        A, Bm, bias, Cf, Cbf, lda, ldb, ldc, M, N, K, zInner, sAo, sAi, sBo,
        sBi, sCo, sCi);
  }
}

static void launch_cvt(hipStream_t stream, const float* src, __bf16* dst,
                       long long n) {
  long long blocks = (n + 255) / 256;
  if (blocks > 4096) blocks = 4096;
  cvt_bf16_kernel<<<(int)blocks, 256, 0, stream>>>(src, dst, n);
}

extern "C" void kernel_launch(void* const* d_in, const int* in_sizes, int n_in,
                              void* d_out, int out_size, void* d_ws,
                              size_t ws_size, hipStream_t stream) {
  (void)in_sizes; (void)n_in; (void)out_size; (void)ws_size;

  const int*   x        = (const int*)d_in[0];
  const int*   amask    = (const int*)d_in[1];
  const float* word_emb = (const float*)d_in[2];
  const float* emb_g    = (const float*)d_in[3];
  const float* emb_b    = (const float*)d_in[4];
  const float* rel_emb  = (const float*)d_in[5];
  const float* Wq  = (const float*)d_in[6];
  const float* bq  = (const float*)d_in[7];
  const float* Wk  = (const float*)d_in[8];
  const float* bk  = (const float*)d_in[9];
  const float* Wv  = (const float*)d_in[10];
  const float* bv  = (const float*)d_in[11];
  const float* Wpk = (const float*)d_in[12];
  const float* bpk = (const float*)d_in[13];
  const float* Wpq = (const float*)d_in[14];
  const float* bpq = (const float*)d_in[15];
  const float* Wo  = (const float*)d_in[16];
  const float* bo  = (const float*)d_in[17];
  const float* ln1g = (const float*)d_in[18];
  const float* ln1b = (const float*)d_in[19];
  const float* W1w  = (const float*)d_in[20];
  const float* b1w  = (const float*)d_in[21];
  const float* W2w  = (const float*)d_in[22];
  const float* b2w  = (const float*)d_in[23];
  const float* ln2g = (const float*)d_in[24];
  const float* ln2b = (const float*)d_in[25];
  const float* clsw = (const float*)d_in[26];
  const float* clsb = (const float*)d_in[27];

  // ---------------- workspace carve (256B aligned bump) ----------------
  char* wsp = (char*)d_ws;
  size_t off = 0;
  auto take = [&](size_t bytes) -> void* {
    off = (off + 255) & ~(size_t)255;
    void* p = wsp + off;
    off += bytes;
    return p;
  };

  const long long WSQ = (long long)HID * HID;  // 262144
  const long long P   = 2 * SPAN;              // 1024

  // Transposed bf16 weights: [l][N][K]
  __bf16* wq_bf  = (__bf16*)take((size_t)NL * WSQ * 2);
  __bf16* wk_bf  = (__bf16*)take((size_t)NL * WSQ * 2);
  __bf16* wv_bf  = (__bf16*)take((size_t)NL * WSQ * 2);
  __bf16* wpk_bf = (__bf16*)take((size_t)NL * WSQ * 2);
  __bf16* wpq_bf = (__bf16*)take((size_t)NL * WSQ * 2);
  __bf16* wo_bf  = (__bf16*)take((size_t)NL * WSQ * 2);
  __bf16* w1_bf  = (__bf16*)take((size_t)NL * HID * FF * 2);  // [l][FF][HID]
  __bf16* w2_bf  = (__bf16*)take((size_t)NL * FF * HID * 2);  // [l][HID][FF]
  __bf16* cls_bf = (__bf16*)take((size_t)VOCAB * HID * 2);    // already [N][K]
  __bf16* rel_bf = (__bf16*)take((size_t)P * HID * 2);        // A operand

  float*  h     = (float*)take((size_t)TOK * HID * 4);
  __bf16* hbf   = (__bf16*)take((size_t)TOK * HID * 2);
  __bf16* qbf   = (__bf16*)take((size_t)TOK * HID * 2);
  __bf16* kbf   = (__bf16*)take((size_t)TOK * HID * 2);
  __bf16* vbf   = (__bf16*)take((size_t)TOK * HID * 2);
  __bf16* vT    = (__bf16*)take((size_t)TOK * HID * 2);       // [b][HID][SEQ]
  __bf16* pkbf  = (__bf16*)take((size_t)P * HID * 2);
  __bf16* pqbf  = (__bf16*)take((size_t)P * HID * 2);
  float*  qk    = (float*)take((size_t)Bn * NHD * SEQ * SEQ * 4);
  float*  qpk   = (float*)take((size_t)Bn * NHD * SEQ * P * 4);
  float*  kpq   = (float*)take((size_t)Bn * NHD * SEQ * P * 4);
  __bf16* probs = (__bf16*)take((size_t)Bn * NHD * SEQ * SEQ * 2);
  __bf16* ctxbf = (__bf16*)take((size_t)TOK * HID * 2);
  float*  attn  = (float*)take((size_t)TOK * HID * 4);
  float*  ff1   = (float*)take((size_t)TOK * FF * 4);
  __bf16* gelubf= (__bf16*)take((size_t)TOK * FF * 2);
  float*  ff2   = (float*)take((size_t)TOK * HID * 4);

  // -------- weight conversion + transposition (every call, deterministic) --
  {
    dim3 gsq(HID / 32, HID / 32, NL);
    transpose_f32_to_bf16<<<gsq, 256, 0, stream>>>(Wq,  wq_bf,  HID, HID);
    transpose_f32_to_bf16<<<gsq, 256, 0, stream>>>(Wk,  wk_bf,  HID, HID);
    transpose_f32_to_bf16<<<gsq, 256, 0, stream>>>(Wv,  wv_bf,  HID, HID);
    transpose_f32_to_bf16<<<gsq, 256, 0, stream>>>(Wpk, wpk_bf, HID, HID);
    transpose_f32_to_bf16<<<gsq, 256, 0, stream>>>(Wpq, wpq_bf, HID, HID);
    transpose_f32_to_bf16<<<gsq, 256, 0, stream>>>(Wo,  wo_bf,  HID, HID);
    dim3 gw1(FF / 32, HID / 32, NL);
    transpose_f32_to_bf16<<<gw1, 256, 0, stream>>>(W1w, w1_bf, HID, FF);
    dim3 gw2(HID / 32, FF / 32, NL);
    transpose_f32_to_bf16<<<gw2, 256, 0, stream>>>(W2w, w2_bf, FF, HID);
  }
  launch_cvt(stream, clsw, cls_bf, (long long)VOCAB * HID);
  launch_cvt(stream, rel_emb, rel_bf, P * HID);

  // ---------------- embedding + LN ----------------
  embed_ln_kernel<<<TOK, 256, 0, stream>>>(x, amask, word_emb, emb_g, emb_b,
                                           h, hbf);

  const long long sBH = (long long)SEQ * HID;  // per-batch stride in q/k/v
  const long long sSS = (long long)SEQ * SEQ;
  const long long sSP = (long long)SEQ * P;

  for (int l = 0; l < NL; ++l) {
    const __bf16* wq  = wq_bf  + (size_t)l * WSQ;
    const __bf16* wk  = wk_bf  + (size_t)l * WSQ;
    const __bf16* wv  = wv_bf  + (size_t)l * WSQ;
    const __bf16* wpk = wpk_bf + (size_t)l * WSQ;
    const __bf16* wpq = wpq_bf + (size_t)l * WSQ;
    const __bf16* wo  = wo_bf  + (size_t)l * WSQ;
    const __bf16* w1  = w1_bf  + (size_t)l * HID * FF;
    const __bf16* w2  = w2_bf  + (size_t)l * FF * HID;

    // q/k/v projections: [2048,512] x [512,512]^T -> bf16
    launch_gemm(stream, hbf, wq, bq + l * HID, nullptr, qbf, HID, HID, HID,
                TOK, HID, HID, 1, 1, 0, 0, 0, 0, 0, 0);
    launch_gemm(stream, hbf, wk, bk + l * HID, nullptr, kbf, HID, HID, HID,
                TOK, HID, HID, 1, 1, 0, 0, 0, 0, 0, 0);
    launch_gemm(stream, hbf, wv, bv + l * HID, nullptr, vbf, HID, HID, HID,
                TOK, HID, HID, 1, 1, 0, 0, 0, 0, 0, 0);
    // relative-position projections: [1024,512] x [512,512]^T -> bf16
    launch_gemm(stream, rel_bf, wpk, bpk + l * HID, nullptr, pkbf, HID, HID,
                HID, (int)P, HID, HID, 1, 1, 0, 0, 0, 0, 0, 0);
    launch_gemm(stream, rel_bf, wpq, bpq + l * HID, nullptr, pqbf, HID, HID,
                HID, (int)P, HID, HID, 1, 1, 0, 0, 0, 0, 0, 0);

    // v^T per batch for the probs@v GEMM (K-contiguous B)
    {
      dim3 gt(HID / 32, SEQ / 32, Bn);
      transpose_bf16<<<gt, 256, 0, stream>>>(vbf, vT, SEQ, HID);
    }

    // qk[b,h] = q_slice @ k_slice^T  (M=512, N=512, K=64), batch = 32
    launch_gemm(stream, qbf, kbf, nullptr, qk, nullptr, HID, HID, SEQ, SEQ,
                SEQ, HEAD, Bn * NHD, NHD, sBH, HEAD, sBH, HEAD,
                (long long)NHD * sSS, sSS);
    // qpk[b,h] = q_slice @ pk_slice^T (M=512, N=1024, K=64)
    launch_gemm(stream, qbf, pkbf, nullptr, qpk, nullptr, HID, HID, (int)P,
                SEQ, (int)P, HEAD, Bn * NHD, NHD, sBH, HEAD, 0, HEAD,
                (long long)NHD * sSP, sSP);
    // kpq[b,h] = k_slice @ pq_slice^T (M=512, N=1024, K=64)
    launch_gemm(stream, kbf, pqbf, nullptr, kpq, nullptr, HID, HID, (int)P,
                SEQ, (int)P, HEAD, Bn * NHD, NHD, sBH, HEAD, 0, HEAD,
                (long long)NHD * sSP, sSP);

    // gather + scale + mask + softmax -> probs (bf16)
    {
      dim3 gs(SEQ, NHD, Bn);
      attn_softmax_kernel<<<gs, 256, 0, stream>>>(qk, qpk, kpq, amask, probs);
    }

    // ctx[b,h] = probs @ v_slice (M=512, N=64, K=512) -> [B,S,HID] bf16
    // B = vT[b][head*64+n][j], ldb = SEQ.
    launch_gemm(stream, probs, vT, nullptr, nullptr, ctxbf, SEQ, SEQ, HID,
                SEQ, HEAD, SEQ, Bn * NHD, NHD, (long long)NHD * sSS, sSS,
                (long long)HID * SEQ, (long long)HEAD * SEQ, sBH, HEAD);

    // attention output projection
    launch_gemm(stream, ctxbf, wo, bo + l * HID, attn, nullptr, HID, HID, HID,
                TOK, HID, HID, 1, 1, 0, 0, 0, 0, 0, 0);
    // h = LN(attn + h)
    residual_ln_kernel<<<TOK, 256, 0, stream>>>(attn, h, ln1g + l * HID,
                                                ln1b + l * HID, h, hbf);

    // FFN: ff1 = h @ W1 + b1 ; gelu ; ff2 = gelu @ W2 + b2
    launch_gemm(stream, hbf, w1, b1w + l * FF, ff1, nullptr, HID, HID, FF,
                TOK, FF, HID, 1, 1, 0, 0, 0, 0, 0, 0);
    gelu_bf16_kernel<<<2048, 256, 0, stream>>>(ff1, gelubf,
                                               (long long)TOK * FF);
    launch_gemm(stream, gelubf, w2, b2w + l * HID, ff2, nullptr, FF, FF, HID,
                TOK, HID, FF, 1, 1, 0, 0, 0, 0, 0, 0);
    // h = LN(ff2 + h)
    residual_ln_kernel<<<TOK, 256, 0, stream>>>(ff2, h, ln2g + l * HID,
                                                ln2b + l * HID, h, hbf);
  }

  // logits = h @ cls_w^T + cls_b  (N=25, B already [N][K], EDGE path)
  float* out_f = (float*)d_out;
  float* logits = out_f + (size_t)TOK * HID;
  launch_gemm(stream, hbf, cls_bf, clsb, logits, nullptr, HID, HID, VOCAB,
              TOK, VOCAB, HID, 1, 1, 0, 0, 0, 0, 0, 0);

  // first output: final hidden states
  hipMemcpyAsync(out_f, h, (size_t)TOK * HID * sizeof(float),
                 hipMemcpyDeviceToDevice, stream);
}
